// RecursiveNeuralKernel_46213848105644
// MI455X (gfx1250) — matrix operand order repaired and checked by reading, not compiled, not run on hardware
//
#include <hip/hip_runtime.h>
#include <hip/hip_bf16.h>
#include <math.h>

typedef __bf16 bf16_t;
typedef __attribute__((ext_vector_type(16))) __bf16 v16bf;
typedef __attribute__((ext_vector_type(8)))  __bf16 v8bf;
typedef __attribute__((ext_vector_type(8)))  float  v8f;

#define FD 1.618f
#define D_DIM 1024
#define B_ROWS 256
#define HALF_D 512
#define HTOT (B_ROWS * D_DIM)   // 262144
#define KDIM 1024               // every WMMA GEMM in this net has K = 1024

// ---------------------------------------------------------------------------
// Elementwise / prep kernels
// ---------------------------------------------------------------------------

__global__ __launch_bounds__(256)
void f32_to_bf16_kernel(const float* __restrict__ src, bf16_t* __restrict__ dst, int n) {
  int i = blockIdx.x * 256 + threadIdx.x;
  if (i < n) dst[i] = (bf16_t)src[i];
}

// rec_in(bf16) = h_new(f32) * mod_strength
__global__ __launch_bounds__(256)
void scale_bf16_kernel(const float* __restrict__ x, const float* __restrict__ mod,
                       bf16_t* __restrict__ y, int n) {
  int i = blockIdx.x * 256 + threadIdx.x;
  if (i < n) y[i] = (bf16_t)(x[i] * mod[0]);
}

// h(bf16) = h_new*(1-iw) + rec_out*iw,  iw = sigmoid(rate*depth)
__global__ __launch_bounds__(256)
void combine_bf16_kernel(const float* __restrict__ hn, const float* __restrict__ rec,
                         const float* __restrict__ rate, int depth,
                         bf16_t* __restrict__ h, int n) {
  int i = blockIdx.x * 256 + threadIdx.x;
  if (i < n) {
    float iw = 1.0f / (1.0f + expf(-rate[0] * (float)depth));
    h[i] = (bf16_t)(hn[i] * (1.0f - iw) + rec[i] * iw);
  }
}

// consciousness head: out[m,j] = sigmoid(sum_k c[m,k]*W2[j,k] + b2[j]), j<3
__global__ __launch_bounds__(256)
void cons_kernel(const float* __restrict__ c, const float* __restrict__ W2,
                 const float* __restrict__ b2, float* __restrict__ out) {
  int idx = blockIdx.x * 256 + threadIdx.x;
  if (idx >= B_ROWS * 3) return;
  int m = idx / 3, j = idx % 3;
  const float* cr = c + (size_t)m * HALF_D;
  const float* wr = W2 + (size_t)j * HALF_D;
  float s = b2[j];
  for (int k = 0; k < HALF_D; ++k) s += cr[k] * wr[k];
  out[idx] = 1.0f / (1.0f + expf(-s));
}

// ---------------------------------------------------------------------------
// Fractal activation helper
// ---------------------------------------------------------------------------

__device__ __forceinline__ float fractal_eval(float v, const float w[5]) {
  const float S[5] = {1.0f / (FD * FD), 1.0f / FD, 1.0f, FD, FD * FD};
  float r = 0.0f;
#pragma unroll
  for (int i = 0; i < 5; ++i) {
    float sx = v * S[i];
    r += w[i] * sinf(sx) * expf(-0.1f * fabsf(sx));
  }
  return r;
}

// ---------------------------------------------------------------------------
// WMMA GEMM: Y[M=256, N] = act(X[256,1024] @ Wb[N,1024]^T + bias[N])
// X is bf16 (produced by a previous kernel's epilogue) -> pure load+wmma loop.
// Each wave owns a 16x32 output tile (2 accumulators) for max grid parallelism
// (16 * N/32 wave-jobs; N=1024 -> 64 blocks of 8 waves).
// Epilogue writes f32 and/or bf16 copies of Y.   ACT: 0=none 1=fractal 2=relu
// ---------------------------------------------------------------------------

template <int ACT>
__global__ __launch_bounds__(256)
void gemm_wmma(const bf16_t* __restrict__ X, const bf16_t* __restrict__ Wb,
               const float* __restrict__ bias, float* __restrict__ Yf,
               bf16_t* __restrict__ Yb, int N, const float* __restrict__ sf) {
  const int lane  = threadIdx.x & 31;
  const int wave  = threadIdx.x >> 5;
  const int wid   = blockIdx.x * 8 + wave;
  const int mtile = wid & 15;        // 256 rows / 16
  const int ngrp  = wid >> 4;        // groups of 32 columns
  const int rif   = lane & 15;       // row/col within fragment
  const int half  = lane >> 4;

  // softmax over the 5 scale factors (tiny, per-thread)
  float w5[5];
  if (ACT == 1) {
    float f0 = sf[0], f1 = sf[1], f2 = sf[2], f3 = sf[3], f4 = sf[4];
    float mx = fmaxf(fmaxf(fmaxf(f0, f1), fmaxf(f2, f3)), f4);
    float e0 = expf(f0 - mx), e1 = expf(f1 - mx), e2 = expf(f2 - mx);
    float e3 = expf(f3 - mx), e4 = expf(f4 - mx);
    float s = e0 + e1 + e2 + e3 + e4;
    w5[0] = e0 / s; w5[1] = e1 / s; w5[2] = e2 / s; w5[3] = e3 / s; w5[4] = e4 / s;
  }

  v8f acc0 = {0,0,0,0,0,0,0,0};
  v8f acc1 = {0,0,0,0,0,0,0,0};

  const int m = mtile * 16 + rif;
  const bf16_t* xrow = X + (size_t)m * KDIM;
  const int nbase = ngrp * 32;
  // B fragment: lane holds column n, a contiguous 16-element K run of W's row n
  const bf16_t* w0 = Wb + (size_t)(nbase + rif) * KDIM;
  const bf16_t* w1 = Wb + (size_t)(nbase + 16 + rif) * KDIM;

#pragma unroll 4
  for (int k0 = 0; k0 < KDIM; k0 += 32) {
    // A fragment: lane covers contiguous K = ka..ka+15 (matches B)
    int ka = k0 + half * 16;
    v8bf alo = *(const v8bf*)(xrow + ka);
    v8bf ahi = *(const v8bf*)(xrow + ka + 8);
    v16bf a;
#pragma unroll
    for (int j = 0; j < 8; ++j) { a[j] = alo[j]; a[8 + j] = ahi[j]; }

    // B fragments: lanes 0-15 K=k0..k0+15, lanes 16-31 K=k0+16..k0+31
    int kb = k0 + half * 16;
    v16bf b0 = *(const v16bf*)(w0 + kb);
    v16bf b1 = *(const v16bf*)(w1 + kb);

    acc0 = __builtin_amdgcn_wmma_f32_16x16x32_bf16(false, a, false, b0, (short)0, acc0, false, false);
    acc1 = __builtin_amdgcn_wmma_f32_16x16x32_bf16(false, a, false, b1, (short)0, acc1, false, false);
  }

  // Epilogue: C/D layout -> VGPR r holds (row = r + 8*half, col = rif)
  float bv0 = bias[nbase + rif];
  float bv1 = bias[nbase + 16 + rif];
#pragma unroll
  for (int r = 0; r < 8; ++r) {
    int mm = mtile * 16 + r + half * 8;
    size_t base = (size_t)mm * N + nbase + rif;
    float v0 = acc0[r] + bv0;
    float v1 = acc1[r] + bv1;
    if (ACT == 1) {
      v0 = fractal_eval(v0, w5);
      v1 = fractal_eval(v1, w5);
    } else if (ACT == 2) {
      v0 = fmaxf(v0, 0.0f);
      v1 = fmaxf(v1, 0.0f);
    }
    if (Yf) { Yf[base] = v0; Yf[base + 16] = v1; }
    if (Yb) { Yb[base] = (bf16_t)v0; Yb[base + 16] = (bf16_t)v1; }
  }
}

// ---------------------------------------------------------------------------
// Host-side orchestration
// ---------------------------------------------------------------------------

namespace {

struct Ctx {
  // device inputs
  const float *b_in, *b_h, *b_out, *b_c1, *W_c2, *b_c2, *sf, *mod, *rate;
  // bf16 weights in ws
  const bf16_t *Wb_in, *Wb_h, *Wb_out, *Wb_c1;
  // per-depth buffers
  bf16_t *h[4], *h2[4], *rin[4];   // bf16 activations (GEMM inputs)
  float  *hn[4], *rout[4];         // f32 values feeding elementwise kernels
  float  *cbuf;                    // 256x512 f32
  hipStream_t stream;
};

inline void launch_gemm(int act, const bf16_t* X, const bf16_t* W, const float* b,
                        float* Yf, bf16_t* Yb, int N, const Ctx& c) {
  dim3 grid((16 * (N / 32)) / 8), block(256);
  if (act == 0)      gemm_wmma<0><<<grid, block, 0, c.stream>>>(X, W, b, Yf, Yb, N, c.sf);
  else if (act == 1) gemm_wmma<1><<<grid, block, 0, c.stream>>>(X, W, b, Yf, Yb, N, c.sf);
  else               gemm_wmma<2><<<grid, block, 0, c.stream>>>(X, W, b, Yf, Yb, N, c.sf);
}

void forward_rec(Ctx& c, const bf16_t* x, float* out, bool top, int depth) {
  const dim3 eg(HTOT / 256), eb(256);
  bf16_t* h = c.h[depth];

  // h = fractal(linear(x, W_in)) -> only consumed by GEMMs, keep bf16 only
  launch_gemm(1, x, c.Wb_in, c.b_in, nullptr, h, D_DIM, c);

  if (depth >= 3) {
    bf16_t* cur = h;
    bf16_t* alt = c.h2[depth];
    for (int i = 0; i < 3; ++i) {
      launch_gemm(1, cur, c.Wb_h + (size_t)i * D_DIM * D_DIM, c.b_h + (size_t)i * D_DIM,
                  nullptr, alt, D_DIM, c);
      bf16_t* t = cur; cur = alt; alt = t;   // ping-pong
    }
    launch_gemm(0, cur, c.Wb_out, c.b_out, out, nullptr, D_DIM, c);
  } else {
    for (int i = 0; i < 3; ++i) {
      // h_new = fractal(linear(h, W_h[i])) -> only feeds scale/combine, keep f32
      launch_gemm(1, h, c.Wb_h + (size_t)i * D_DIM * D_DIM, c.b_h + (size_t)i * D_DIM,
                  c.hn[depth], nullptr, D_DIM, c);
      // recursive input (bf16) = h_new * mod_strength
      scale_bf16_kernel<<<eg, eb, 0, c.stream>>>(c.hn[depth], c.mod, c.rin[depth], HTOT);
      forward_rec(c, c.rin[depth], c.rout[depth], false, depth + 1);
      // h(bf16) = h_new*(1-iw) + rec_out*iw
      combine_bf16_kernel<<<eg, eb, 0, c.stream>>>(c.hn[depth], c.rout[depth], c.rate,
                                                   depth, h, HTOT);
    }
    launch_gemm(0, h, c.Wb_out, c.b_out, out, nullptr, D_DIM, c);
    if (top) {
      // consciousness head (sub-call heads are discarded by the reference)
      launch_gemm(2, h, c.Wb_c1, c.b_c1, c.cbuf, nullptr, HALF_D, c);
      cons_kernel<<<3, 256, 0, c.stream>>>(c.cbuf, c.W_c2, c.b_c2, out + (size_t)HTOT);
    }
  }
}

}  // namespace

extern "C" void kernel_launch(void* const* d_in, const int* in_sizes, int n_in,
                              void* d_out, int out_size, void* d_ws, size_t ws_size,
                              hipStream_t stream) {
  (void)in_sizes; (void)n_in; (void)out_size; (void)ws_size;

  const float* x     = (const float*)d_in[0];
  const float* W_in  = (const float*)d_in[1];
  const float* b_in  = (const float*)d_in[2];
  const float* W_h   = (const float*)d_in[3];
  const float* b_h   = (const float*)d_in[4];
  const float* W_out = (const float*)d_in[5];
  const float* b_out = (const float*)d_in[6];
  const float* W_c1  = (const float*)d_in[7];
  const float* b_c1  = (const float*)d_in[8];
  const float* W_c2  = (const float*)d_in[9];
  const float* b_c2  = (const float*)d_in[10];
  const float* sf    = (const float*)d_in[11];
  const float* mod   = (const float*)d_in[12];
  const float* rate  = (const float*)d_in[13];

  // carve workspace (~26 MB)
  char* p = (char*)d_ws;
  bf16_t* Wb_in  = (bf16_t*)p; p += (size_t)D_DIM * D_DIM * 2;
  bf16_t* Wb_h   = (bf16_t*)p; p += (size_t)3 * D_DIM * D_DIM * 2;
  bf16_t* Wb_out = (bf16_t*)p; p += (size_t)D_DIM * D_DIM * 2;
  bf16_t* Wb_c1  = (bf16_t*)p; p += (size_t)HALF_D * D_DIM * 2;
  bf16_t* xb     = (bf16_t*)p; p += (size_t)HTOT * 2;

  Ctx c;
  for (int d = 0; d < 4; ++d) {
    c.h[d]   = (bf16_t*)p; p += (size_t)HTOT * 2;
    c.h2[d]  = (bf16_t*)p; p += (size_t)HTOT * 2;
    c.rin[d] = (bf16_t*)p; p += (size_t)HTOT * 2;
    c.hn[d]   = (float*)p; p += (size_t)HTOT * 4;
    c.rout[d] = (float*)p; p += (size_t)HTOT * 4;
  }
  c.cbuf = (float*)p; p += (size_t)B_ROWS * HALF_D * 4;

  // convert weights + top-level input to bf16 (L2-resident, reused 40x)
  {
    int n;
    n = D_DIM * D_DIM;
    f32_to_bf16_kernel<<<(n + 255) / 256, 256, 0, stream>>>(W_in, Wb_in, n);
    n = 3 * D_DIM * D_DIM;
    f32_to_bf16_kernel<<<(n + 255) / 256, 256, 0, stream>>>(W_h, Wb_h, n);
    n = D_DIM * D_DIM;
    f32_to_bf16_kernel<<<(n + 255) / 256, 256, 0, stream>>>(W_out, Wb_out, n);
    n = HALF_D * D_DIM;
    f32_to_bf16_kernel<<<(n + 255) / 256, 256, 0, stream>>>(W_c1, Wb_c1, n);
    n = HTOT;
    f32_to_bf16_kernel<<<(n + 255) / 256, 256, 0, stream>>>(x, xb, n);
  }

  c.b_in = b_in; c.b_h = b_h; c.b_out = b_out; c.b_c1 = b_c1;
  c.W_c2 = W_c2; c.b_c2 = b_c2; c.sf = sf; c.mod = mod; c.rate = rate;
  c.Wb_in = Wb_in; c.Wb_h = Wb_h; c.Wb_out = Wb_out; c.Wb_c1 = Wb_c1;
  c.stream = stream;

  forward_rec(c, xb, (float*)d_out, true, 0);
}
